// AtariAgent57LitePolicy_90520730731002
// MI455X (gfx1250) — matrix operand order repaired and checked by reading, not compiled, not run on hardware
//
#include <hip/hip_runtime.h>
#include <hip/hip_bf16.h>
#include <math.h>

// ---------------- problem constants ----------------
#define BB 1024
#define CIN 4
#define IMG 84
#define AACT 18
#define HH 32
#define CONV_DIM 392            // 8*7*7
#define KFEAT 416               // 392 padded to mult of 32
#define KCOMB 96                // 32 proj + 18 one-hot + 1 reward + 32 h + 13 zero pad
#define NGATE 128

#define CDIV(a, b) (((a) + (b) - 1) / (b))

typedef __attribute__((ext_vector_type(16))) __bf16 v16bf;
typedef __attribute__((ext_vector_type(8)))  __bf16 bf16x8;
typedef __attribute__((ext_vector_type(8)))  float  v8f;

static __device__ __forceinline__ __bf16 f2bf(float f) {
  unsigned u = __builtin_bit_cast(unsigned, f);
  unsigned r = u + 0x7FFFu + ((u >> 16) & 1u);   // round-to-nearest-even
  unsigned short h = (unsigned short)(r >> 16);
  return __builtin_bit_cast(__bf16, h);
}

static __device__ __forceinline__ float sigmoidf_(float x) {
  return 1.0f / (1.0f + __expf(-x));
}

// ---------------- conv1: [B,4,84,84] -> relu -> [B,4,20,20], k8 s4 ----------------
__global__ void conv1_relu_k(const float* __restrict__ x, const float* __restrict__ w,
                             const float* __restrict__ bias, float* __restrict__ y) {
  int idx = blockIdx.x * blockDim.x + threadIdx.x;
  if (idx >= BB * 4 * 20 * 20) return;
  int ox = idx % 20; int t = idx / 20;
  int oy = t % 20;   t /= 20;
  int oc = t % 4;    int b = t / 4;
  const float* xb = x + (size_t)b * (CIN * IMG * IMG);
  float acc = bias[oc];
#pragma unroll
  for (int ic = 0; ic < 4; ++ic) {
    const float* xc = xb + ic * IMG * IMG + (oy * 4) * IMG + ox * 4;
    const float* wc = w + ((oc * 4 + ic) * 8) * 8;
    __builtin_prefetch(xc + 8 * IMG, 0, 1);     // global_prefetch_b8
#pragma unroll
    for (int ky = 0; ky < 8; ++ky)
#pragma unroll
      for (int kx = 0; kx < 8; ++kx)
        acc = fmaf(xc[ky * IMG + kx], wc[ky * 8 + kx], acc);
  }
  y[idx] = fmaxf(acc, 0.0f);
}

// ---------------- conv2: [B,4,20,20] -> relu -> [B,8,9,9], k4 s2 ----------------
__global__ void conv2_relu_k(const float* __restrict__ x, const float* __restrict__ w,
                             const float* __restrict__ bias, float* __restrict__ y) {
  int idx = blockIdx.x * blockDim.x + threadIdx.x;
  if (idx >= BB * 8 * 9 * 9) return;
  int ox = idx % 9; int t = idx / 9;
  int oy = t % 9;   t /= 9;
  int oc = t % 8;   int b = t / 8;
  const float* xb = x + (size_t)b * (4 * 20 * 20);
  float acc = bias[oc];
#pragma unroll
  for (int ic = 0; ic < 4; ++ic) {
    const float* xc = xb + ic * 400 + (oy * 2) * 20 + ox * 2;
    const float* wc = w + ((oc * 4 + ic) * 4) * 4;
#pragma unroll
    for (int ky = 0; ky < 4; ++ky)
#pragma unroll
      for (int kx = 0; kx < 4; ++kx)
        acc = fmaf(xc[ky * 20 + kx], wc[ky * 4 + kx], acc);
  }
  y[idx] = fmaxf(acc, 0.0f);
}

// ------- conv3: [B,8,9,9] -> relu -> bf16 feat_bf [B,416] (cols>=392 padded elsewhere) -------
__global__ void conv3_relu_k(const float* __restrict__ x, const float* __restrict__ w,
                             const float* __restrict__ bias, __bf16* __restrict__ feat_bf) {
  int idx = blockIdx.x * blockDim.x + threadIdx.x;
  if (idx >= BB * 8 * 7 * 7) return;
  int ox = idx % 7; int t = idx / 7;
  int oy = t % 7;   t /= 7;
  int oc = t % 8;   int b = t / 8;
  const float* xb = x + (size_t)b * (8 * 81);
  float acc = bias[oc];
#pragma unroll
  for (int ic = 0; ic < 8; ++ic) {
    const float* xc = xb + ic * 81 + oy * 9 + ox;
    const float* wc = w + ((oc * 8 + ic) * 3) * 3;
#pragma unroll
    for (int ky = 0; ky < 3; ++ky)
#pragma unroll
      for (int kx = 0; kx < 3; ++kx)
        acc = fmaf(xc[ky * 9 + kx], wc[ky * 3 + kx], acc);
  }
  feat_bf[(size_t)b * KFEAT + oc * 49 + oy * 7 + ox] = f2bf(fmaxf(acc, 0.0f));
}

// ---------------- zero the K padding columns of feat_bf (cols 392..415) ----------------
__global__ void pad_feat_k(__bf16* __restrict__ feat_bf) {
  int idx = blockIdx.x * blockDim.x + threadIdx.x;
  const int pad = KFEAT - CONV_DIM;  // 24
  if (idx >= BB * pad) return;
  int b = idx / pad, c = CONV_DIM + idx % pad;
  feat_bf[(size_t)b * KFEAT + c] = f2bf(0.0f);
}

// ---------------- pack proj_w [32,392] f32 -> [32,416] bf16 (zero-padded) ----------------
__global__ void pack_projw_k(const float* __restrict__ w, __bf16* __restrict__ wp) {
  int idx = blockIdx.x * blockDim.x + threadIdx.x;
  if (idx >= 32 * KFEAT) return;
  int k = idx % KFEAT, n = idx / KFEAT;
  wp[idx] = f2bf(k < CONV_DIM ? w[n * CONV_DIM + k] : 0.0f);
}

// ---- acts_bf[:,32:96] = one-hot(action) | prev_reward | h | zeros (bf16) ----
__global__ void fill_acts_k(const float* __restrict__ h, const int* __restrict__ pa,
                            const float* __restrict__ pr, __bf16* __restrict__ acts_bf) {
  int idx = blockIdx.x * blockDim.x + threadIdx.x;
  const int cols = KCOMB - 32;  // 64
  if (idx >= BB * cols) return;
  int col = 32 + idx % cols;
  int b = idx / cols;
  int a = pa[0]; a = a < 0 ? 0 : (a > AACT - 1 ? AACT - 1 : a);
  float v;
  if (col < 32 + AACT)  v = (col - 32 == a) ? 1.0f : 0.0f;
  else if (col == 50)   v = pr[0];
  else if (col < 83)    v = h[b * HH + (col - 51)];
  else                  v = 0.0f;
  acts_bf[(size_t)b * KCOMB + col] = f2bf(v);
}

// ---- combined LSTM weight bf16 [128,96] = [W_ih | W_hh | 0], bias = b_ih + b_hh ----
__global__ void pack_wcomb_k(const float* __restrict__ wih, const float* __restrict__ whh,
                             const float* __restrict__ bih, const float* __restrict__ bhh,
                             __bf16* __restrict__ wcomb, float* __restrict__ bcomb) {
  int idx = blockIdx.x * blockDim.x + threadIdx.x;
  if (idx >= NGATE * KCOMB) return;
  int k = idx % KCOMB, n = idx / KCOMB;
  float v = (k < 51) ? wih[n * 51 + k] : ((k < 83) ? whh[n * 32 + (k - 51)] : 0.0f);
  wcomb[idx] = f2bf(v);
  if (k == 0) bcomb[n] = bih[n] + bhh[n];
}

// ---- q_w [18,32] f32 -> [32,32] bf16 (rows 18..31 zero) ----
__global__ void pack_qw_k(const float* __restrict__ w, __bf16* __restrict__ wp) {
  int idx = blockIdx.x * blockDim.x + threadIdx.x;
  if (idx >= 32 * HH) return;
  int k = idx % HH, n = idx / HH;
  wp[idx] = f2bf(n < AACT ? w[n * HH + k] : 0.0f);
}

// ------------- WMMA bf16 GEMM: out[M,N] = A[M,KPAD]bf16 @ W[Npad,KPAD]bf16^T + bias -------------
// Pre-padded operands: no bounds checks in the K loop. One wave per 16x16 tile.
// grid = (M/16, Npad/16), block = 32. KPAD compile-time -> fully unrolled (KPAD/32 wmma ops).
template <int KPAD, bool OUT_BF16>
__global__ void gemm_wmma_k(const __bf16* __restrict__ A, const __bf16* __restrict__ W,
                            const float* __restrict__ bias, void* __restrict__ outp,
                            int Nreal, int ldo) {
  const int lane = threadIdx.x & 31;
  const int half = lane >> 4;
  const int l16  = lane & 15;
  const int m0 = blockIdx.x * 16;
  const int n0 = blockIdx.y * 16;

  const __bf16* arow = A + (size_t)(m0 + l16) * KPAD;   // A fragment rows: M = l16
  const __bf16* wrow = W + (size_t)(n0 + l16) * KPAD;   // B fragment cols: N = l16
  v8f acc = {};

#pragma unroll
  for (int k0 = 0; k0 < KPAD; k0 += 32) {
    // A 16x32 fragment: lane needs K = k0 + {8*half+0..7} and {16+8*half+0..7}
    bf16x8 a_lo = *(const bf16x8*)(arow + k0 + 8 * half);
    bf16x8 a_hi = *(const bf16x8*)(arow + k0 + 16 + 8 * half);
    // B 32x16 fragment: lane needs K = k0 + {8*half+0..7} and {16+8*half+0..7}
    bf16x8 b_lo = *(const bf16x8*)(wrow + k0 + 8 * half);
    bf16x8 b_hi = *(const bf16x8*)(wrow + k0 + 16 + 8 * half);
    v16bf a = __builtin_shufflevector(a_lo, a_hi, 0, 1, 2, 3, 4, 5, 6, 7,
                                      8, 9, 10, 11, 12, 13, 14, 15);
    v16bf b = __builtin_shufflevector(b_lo, b_hi, 0, 1, 2, 3, 4, 5, 6, 7,
                                      8, 9, 10, 11, 12, 13, 14, 15);
    acc = __builtin_amdgcn_wmma_f32_16x16x32_bf16(
        false, a, false, b, (short)0, acc, false, false);
  }

#pragma unroll
  for (int v = 0; v < 8; ++v) {
    int row = m0 + v + 8 * half;   // C/D layout: VGPR v, lane half
    int col = n0 + l16;
    if (col < Nreal) {
      float r = acc[v] + bias[col];
      if (OUT_BF16) ((__bf16*)outp)[(size_t)row * ldo + col] = f2bf(r);
      else          ((float*)outp)[(size_t)row * ldo + col] = r;
    }
  }
}

// ---------------- LSTM elementwise: gates [B,128] (i,f,g,o) -> h_new, c_new ----------------
__global__ void lstm_elem_k(const float* __restrict__ gates, const float* __restrict__ c_in,
                            float* __restrict__ h_out, float* __restrict__ c_out,
                            __bf16* __restrict__ h_bf) {
  int idx = blockIdx.x * blockDim.x + threadIdx.x;
  if (idx >= BB * HH) return;
  int b = idx >> 5, j = idx & 31;
  float ig = gates[b * NGATE + j];
  float fg = gates[b * NGATE + 32 + j];
  float gg = gates[b * NGATE + 64 + j];
  float og = gates[b * NGATE + 96 + j];
  float cn = sigmoidf_(fg) * c_in[idx] + sigmoidf_(ig) * tanhf(gg);
  float hn = sigmoidf_(og) * tanhf(cn);
  h_out[idx] = hn;
  c_out[idx] = cn;
  h_bf[idx]  = f2bf(hn);
}

// ---------------- host launcher ----------------
extern "C" void kernel_launch(void* const* d_in, const int* in_sizes, int n_in,
                              void* d_out, int out_size, void* d_ws, size_t ws_size,
                              hipStream_t stream) {
  const float* x     = (const float*)d_in[0];
  const int*   pa    = (const int*)  d_in[1];
  const float* pr    = (const float*)d_in[2];
  const float* h_in  = (const float*)d_in[3];
  const float* c_in  = (const float*)d_in[4];
  const float* c1w   = (const float*)d_in[5];
  const float* c1b   = (const float*)d_in[6];
  const float* c2w   = (const float*)d_in[7];
  const float* c2b   = (const float*)d_in[8];
  const float* c3w   = (const float*)d_in[9];
  const float* c3b   = (const float*)d_in[10];
  const float* projw = (const float*)d_in[11];
  const float* projb = (const float*)d_in[12];
  const float* wih   = (const float*)d_in[13];
  const float* whh   = (const float*)d_in[14];
  const float* bih   = (const float*)d_in[15];
  const float* bhh   = (const float*)d_in[16];
  const float* qw    = (const float*)d_in[17];
  const float* qb    = (const float*)d_in[18];
  float* out = (float*)d_out;

  // ---- workspace carve-up (all blocks kept 256B-aligned) ----
  char* p = (char*)d_ws;
  float*  y1       = (float*)p;  p += (size_t)BB * 4 * 400 * 4;      // 6,553,600
  float*  y2       = (float*)p;  p += (size_t)BB * 8 * 81 * 4;       // 2,654,208
  float*  gates    = (float*)p;  p += (size_t)BB * NGATE * 4;        //   524,288
  float*  bcomb    = (float*)p;  p += 256;                           //   128 f32
  __bf16* feat_bf  = (__bf16*)p; p += (size_t)BB * KFEAT * 2;        //   851,968
  __bf16* projw_bf = (__bf16*)p; p += (size_t)32 * KFEAT * 2;        //    26,624
  __bf16* acts_bf  = (__bf16*)p; p += (size_t)BB * KCOMB * 2;        //   196,608
  __bf16* wcomb_bf = (__bf16*)p; p += (size_t)NGATE * KCOMB * 2;     //    24,576
  __bf16* qw_bf    = (__bf16*)p; p += (size_t)32 * HH * 2;           //     2,048
  __bf16* h_bf     = (__bf16*)p; p += (size_t)BB * HH * 2;           //    65,536

  float* q_out = out;                       // [B,18]
  float* h_out = out + (size_t)BB * AACT;   // [B,32]
  float* c_out = h_out + (size_t)BB * HH;   // [B,32]

  const int blk = 256;

  // CNN encoder (memory-bound streaming; x = 115.6 MB is the roofline term)
  conv1_relu_k<<<CDIV(BB * 4 * 400, blk), blk, 0, stream>>>(x, c1w, c1b, y1);
  conv2_relu_k<<<CDIV(BB * 8 * 81, blk), blk, 0, stream>>>(y1, c2w, c2b, y2);
  conv3_relu_k<<<CDIV(BB * 8 * 49, blk), blk, 0, stream>>>(y2, c3w, c3b, feat_bf);
  pad_feat_k<<<CDIV(BB * (KFEAT - CONV_DIM), blk), blk, 0, stream>>>(feat_bf);

  // weight packing (tiny, once per launch, deterministic)
  pack_projw_k<<<CDIV(32 * KFEAT, blk), blk, 0, stream>>>(projw, projw_bf);
  pack_wcomb_k<<<CDIV(NGATE * KCOMB, blk), blk, 0, stream>>>(wih, whh, bih, bhh,
                                                             wcomb_bf, bcomb);
  pack_qw_k<<<CDIV(32 * HH, blk), blk, 0, stream>>>(qw, qw_bf);

  // proj GEMM (WMMA, K=416): acts_bf[:,0:32] = bf16(feat @ proj_w^T + proj_b)
  {
    dim3 grid(BB / 16, 2);
    gemm_wmma_k<KFEAT, true><<<grid, 32, 0, stream>>>(feat_bf, projw_bf, projb,
                                                      acts_bf, 32, KCOMB);
  }
  // acts_bf[:,32:96] = one-hot | reward | h | 0
  fill_acts_k<<<CDIV(BB * (KCOMB - 32), blk), blk, 0, stream>>>(h_in, pa, pr, acts_bf);

  // gates GEMM (WMMA, K=96): gates = acts @ [W_ih|W_hh]^T + (b_ih+b_hh)
  {
    dim3 grid(BB / 16, NGATE / 16);
    gemm_wmma_k<KCOMB, false><<<grid, 32, 0, stream>>>(acts_bf, wcomb_bf, bcomb,
                                                       gates, NGATE, NGATE);
  }

  // LSTM cell elementwise -> h_new, c_new (+ bf16 copy of h_new for Q head)
  lstm_elem_k<<<CDIV(BB * HH, blk), blk, 0, stream>>>(gates, c_in, h_out, c_out, h_bf);

  // q GEMM (WMMA, K=32): q = h_new @ q_w^T + q_b  (N padded to 32, store guarded to 18)
  {
    dim3 grid(BB / 16, 2);
    gemm_wmma_k<HH, false><<<grid, 32, 0, stream>>>(h_bf, qw_bf, qb,
                                                    q_out, AACT, AACT);
  }
}
